// pyramid_step_model_41824391528690
// MI455X (gfx1250) — matrix operand order repaired and checked by reading, not compiled, not run on hardware
//
#include <hip/hip_runtime.h>
#include <hip/hip_bf16.h>

#define B_N 8
#define T_N 8192
#define S_N 8192
#define NH_N 8
#define MD_N 32
#define FF_N 1024
#define FFNH_N 128
#define H_N 4
#define LOG100F 4.6051701859880914f

typedef __bf16 bf16_t;
typedef bf16_t v16bf __attribute__((ext_vector_type(16)));
typedef float v8f __attribute__((ext_vector_type(8)));
typedef unsigned int uv4 __attribute__((ext_vector_type(4)));

union FragBF { v16bf v; uv4 q[2]; };

// ---------------------------------------------------------------- kNN top-8
__global__ void __launch_bounds__(256) knn_kernel(const float* __restrict__ ct,
                                                  const float* __restrict__ cs,
                                                  int* __restrict__ idx_out) {
  int t = blockIdx.x * blockDim.x + threadIdx.x;
  float tx = ct[t];
  float ty = ct[T_N + t];
  float t2 = tx * tx + ty * ty;
  float bd[NH_N];
  int bi[NH_N];
#pragma unroll
  for (int j = 0; j < NH_N; ++j) { bd[j] = 3.0e38f; bi[j] = 0; }
  for (int s = 0; s < S_N; ++s) {          // s is wave-uniform -> scalar loads
    float sx = cs[s];
    float sy = cs[S_N + s];
    float d = t2 + sx * sx + sy * sy - 2.0f * (tx * sx + ty * sy);
    if (d < bd[NH_N - 1]) {                // strict <: earlier index wins ties
      bd[NH_N - 1] = d; bi[NH_N - 1] = s;
#pragma unroll
      for (int j = NH_N - 1; j > 0; --j) {
        if (bd[j] < bd[j - 1]) {
          float td = bd[j]; bd[j] = bd[j - 1]; bd[j - 1] = td;
          int ti = bi[j]; bi[j] = bi[j - 1]; bi[j - 1] = ti;
        }
      }
    }
  }
#pragma unroll
  for (int j = 0; j < NH_N; ++j) idx_out[t * NH_N + j] = bi[j];
}

// --------------------------------------------------------- h = MLP(x) 3->32->32
__global__ void __launch_bounds__(256) embed_kernel(
    const float* __restrict__ x, const float* __restrict__ w1,
    const float* __restrict__ b1, const float* __restrict__ w2,
    const float* __restrict__ b2, float* __restrict__ h) {
  __shared__ float sw1[3 * MD_N];
  __shared__ float sb1[MD_N];
  __shared__ float sw2[MD_N * MD_N];
  __shared__ float sb2[MD_N];
  for (int i = threadIdx.x; i < MD_N * MD_N; i += 256) sw2[i] = w2[i];
  if (threadIdx.x < 3 * MD_N) sw1[threadIdx.x] = w1[threadIdx.x];
  if (threadIdx.x < MD_N) { sb1[threadIdx.x] = b1[threadIdx.x]; sb2[threadIdx.x] = b2[threadIdx.x]; }
  __syncthreads();
  size_t r = (size_t)blockIdx.x * blockDim.x + threadIdx.x;   // [0, B*S)
  float x0 = x[r * 3 + 0], x1 = x[r * 3 + 1], x2 = x[r * 3 + 2];
  float t1[MD_N];
#pragma unroll
  for (int c = 0; c < MD_N; ++c) {
    float a = x0 * sw1[c] + x1 * sw1[MD_N + c] + x2 * sw1[2 * MD_N + c] + sb1[c];
    t1[c] = a > 0.f ? a : 0.2f * a;
  }
  float* hr = h + r * MD_N;
  for (int c = 0; c < MD_N; ++c) {
    float a = sb2[c];
#pragma unroll
    for (int mm = 0; mm < MD_N; ++mm) a += t1[mm] * sw2[mm * MD_N + c];
    hr[c] = a;
  }
}

// -------------------------------------------------- pe = MLP(rel) 2->128->32
__global__ void __launch_bounds__(256) pe_kernel(
    const float* __restrict__ ct, const float* __restrict__ cs,
    const int* __restrict__ idx, const float* __restrict__ w1,
    const float* __restrict__ b1, const float* __restrict__ w2,
    const float* __restrict__ b2, float* __restrict__ pe) {
  __shared__ float sw1[2 * FFNH_N];
  __shared__ float sb1[FFNH_N];
  __shared__ float sw2[FFNH_N * MD_N];
  __shared__ float sb2[MD_N];
  for (int i = threadIdx.x; i < FFNH_N * MD_N; i += 256) sw2[i] = w2[i];
  if (threadIdx.x < 2 * FFNH_N) sw1[threadIdx.x] = w1[threadIdx.x];
  if (threadIdx.x < FFNH_N) sb1[threadIdx.x] = b1[threadIdx.x];
  if (threadIdx.x < MD_N) sb2[threadIdx.x] = b2[threadIdx.x];
  __syncthreads();
  int r = blockIdx.x * blockDim.x + threadIdx.x;   // [0, T*NH)
  int t = r >> 3;
  int s = idx[r];
  float rx = cs[s] - ct[t];
  float ry = cs[S_N + s] - ct[T_N + t];
  float acc[MD_N];
#pragma unroll
  for (int c = 0; c < MD_N; ++c) acc[c] = sb2[c];
  for (int mm = 0; mm < FFNH_N; ++mm) {
    float hv = rx * sw1[mm] + ry * sw1[FFNH_N + mm] + sb1[mm];
    hv = hv > 0.f ? hv : 0.2f * hv;
#pragma unroll
    for (int c = 0; c < MD_N; ++c) acc[c] += hv * sw2[mm * MD_N + c];
  }
  float* pr = pe + (size_t)r * MD_N;
#pragma unroll
  for (int c = 0; c < MD_N; ++c) pr[c] = acc[c];
}

// ------------------------------------------- pack mo_w1 into WMMA-B fragments
// packed index: (((kt*64 + nt)*32 + lane)*16 + el)
// lane 0..15: N = nt*16+lane, K = kt*32 + {0..7, 16..23}[el]     (ISA 05_wmma B layout)
// lane 16..31: N = nt*16+lane-16, K = kt*32 + {8..15, 24..31}[el]
__global__ void __launch_bounds__(256) pack_w1_kernel(const float* __restrict__ w1,
                                                      bf16_t* __restrict__ w1p) {
  int e = blockIdx.x * blockDim.x + threadIdx.x;   // [0, 256*1024)
  int el = e & 15;
  int lane = (e >> 4) & 31;
  int tile = e >> 9;
  int nt = tile & 63;
  int kt = tile >> 6;
  int n = nt * 16 + (lane & 15);
  int k = kt * 32 + ((lane >> 4) << 3) + (el & 7) + ((el >> 3) << 4);
  w1p[e] = (bf16_t)w1[(size_t)k * FF_N + n];
}

// ------------- attention: wave per (b,t); lane = (neighbor i, head quadrant)
__global__ void __launch_bounds__(256) attn_kernel(
    const float* __restrict__ h, const float* __restrict__ pe,
    const int* __restrict__ idx,
    const float* __restrict__ qw, const float* __restrict__ kw,
    const float* __restrict__ vw,
    const float* __restrict__ lng, const float* __restrict__ lnb,
    const float* __restrict__ lscale, const float* __restrict__ ow,
    bf16_t* __restrict__ xn) {
  __shared__ float s_qw[MD_N * MD_N];
  __shared__ float s_kw[MD_N * MD_N];
  __shared__ float s_vw[MD_N * MD_N];
  __shared__ float s_ow[MD_N * MD_N];
  __shared__ float s_lg[MD_N];
  __shared__ float s_lb[MD_N];
  __shared__ float s_sc[H_N];
  __shared__ float s_hn[8][NH_N][MD_N];
  __shared__ float s_kn[8][NH_N][MD_N];
  __shared__ float s_vv[8][NH_N][MD_N];
  __shared__ float s_ao[8][NH_N][MD_N];

  for (int i = threadIdx.x; i < MD_N * MD_N; i += 256) {
    s_qw[i] = qw[i]; s_kw[i] = kw[i]; s_vw[i] = vw[i]; s_ow[i] = ow[i];
  }
  if (threadIdx.x < MD_N) { s_lg[threadIdx.x] = lng[threadIdx.x]; s_lb[threadIdx.x] = lnb[threadIdx.x]; }
  if (threadIdx.x < H_N) s_sc[threadIdx.x] = expf(fminf(lscale[threadIdx.x], LOG100F));
  __syncthreads();

  int wave = threadIdx.x >> 5;
  int lane = threadIdx.x & 31;
  int m = blockIdx.x * 8 + wave;        // row in [0, B*T)
  int b = m >> 13;
  int t = m & (T_N - 1);
  int i = lane >> 2;                    // neighbor 0..7
  int q4 = lane & 3;                    // head 0..3 == channel quadrant
  int cbase = q4 * 8;

  int s = idx[t * NH_N + i];
  const float* hrow = h + ((size_t)b * S_N + s) * MD_N;
  const float* prow = pe + ((size_t)t * NH_N + i) * MD_N;

  float xnh[8], z[8];
  float s1 = 0.f, s2 = 0.f;
#pragma unroll
  for (int d = 0; d < 8; ++d) {
    int c = cbase + d;
    xnh[d] = hrow[c];
    z[d] = xnh[d] + prow[c];
    s1 += z[d];
    s2 += z[d] * z[d];
  }
  s1 += __shfl_xor(s1, 1); s1 += __shfl_xor(s1, 2);   // quad = one neighbor
  s2 += __shfl_xor(s2, 1); s2 += __shfl_xor(s2, 2);
  float mu = s1 * (1.0f / MD_N);
  float var = s2 * (1.0f / MD_N) - mu * mu;
  float rstd = rsqrtf(var + 1e-5f);
#pragma unroll
  for (int d = 0; d < 8; ++d) {
    int c = cbase + d;
    s_hn[wave][i][c] = (z[d] - mu) * rstd * s_lg[c] + s_lb[c];
  }
  __syncthreads();

  float qv[8], kv[8], vv[8];
#pragma unroll
  for (int d = 0; d < 8; ++d) {
    int c = cbase + d;
    float aq = 0.f, ak = 0.f, av = 0.f;
    for (int mm = 0; mm < MD_N; ++mm) {
      float hv = s_hn[wave][i][mm];
      aq += hv * s_qw[mm * MD_N + c];
      ak += hv * s_kw[mm * MD_N + c];
      av += hv * s_vw[mm * MD_N + c];
    }
    qv[d] = aq; kv[d] = ak; vv[d] = av;
  }
  float qs = 0.f, ks = 0.f;
#pragma unroll
  for (int d = 0; d < 8; ++d) { qs += qv[d] * qv[d]; ks += kv[d] * kv[d]; }
  float qi = 1.0f / (sqrtf(qs) + 1e-8f);
  float ki = 1.0f / (sqrtf(ks) + 1e-8f);
#pragma unroll
  for (int d = 0; d < 8; ++d) {
    qv[d] *= qi;
    s_kn[wave][i][cbase + d] = kv[d] * ki;
    s_vv[wave][i][cbase + d] = vv[d];
  }
  __syncthreads();

  float sc = s_sc[q4];
  float lg[NH_N];
  float mx = -3.0e38f;
#pragma unroll
  for (int j = 0; j < NH_N; ++j) {
    float a = 0.f;
#pragma unroll
    for (int d = 0; d < 8; ++d) a += qv[d] * s_kn[wave][j][cbase + d];
    lg[j] = a * sc;
    mx = fmaxf(mx, lg[j]);
  }
  float den = 0.f;
#pragma unroll
  for (int j = 0; j < NH_N; ++j) { lg[j] = expf(lg[j] - mx); den += lg[j]; }
  float di = 1.0f / den;
  float ao[8] = {0, 0, 0, 0, 0, 0, 0, 0};
#pragma unroll
  for (int j = 0; j < NH_N; ++j) {
    float wgt = lg[j] * di;
#pragma unroll
    for (int d = 0; d < 8; ++d) ao[d] += wgt * s_vv[wave][j][cbase + d];
  }
#pragma unroll
  for (int d = 0; d < 8; ++d) s_ao[wave][i][cbase + d] = ao[d];
  __syncthreads();

  bf16_t* xrow = xn + (size_t)m * (NH_N * MD_N);
#pragma unroll
  for (int d = 0; d < 8; ++d) {
    int c = cbase + d;
    float a = 0.f;
    for (int mm = 0; mm < MD_N; ++mm) a += s_ao[wave][i][mm] * s_ow[mm * MD_N + c];
    xrow[i * MD_N + c] = (bf16_t)((xnh[d] + a) * 0.5f);
  }
}

// ----------- big MLP: (16 x 256) @ (256 x 1024) WMMA, then 1024->3 reduction
// Per k-step: batch-load all 8 B fragments, then fire 8 back-to-back WMMAs.
// One s_wait per k-step instead of one per WMMA; next k-slab prefetched.
__global__ void __launch_bounds__(256) mlp_kernel(
    const bf16_t* __restrict__ xn, const bf16_t* __restrict__ w1p,
    const float* __restrict__ b1, const float* __restrict__ w2,
    const float* __restrict__ b2, float* __restrict__ out) {
  __shared__ __align__(16) bf16_t sA[16 * 256];     // 8 KB  A tile
  __shared__ __align__(16) bf16_t sFF[16 * FF_N];   // 32 KB lrelu'd FF tile

  int m0 = blockIdx.x * 16;
  {
    const uv4* src = (const uv4*)(xn + (size_t)m0 * 256);
    uv4* dst = (uv4*)sA;
    for (int i = threadIdx.x; i < 512; i += 256) dst[i] = src[i];
  }
  __syncthreads();

  int wave = threadIdx.x >> 5;
  int lane = threadIdx.x & 31;
  int row = lane & 15;
  int khalf = (lane >> 4) << 3;   // A frag: lanes 16-31 start at K+8
  int rbase = (lane >> 4) << 3;   // C frag: lanes 16-31 hold M=8..15

  v8f acc[8] = {};
#pragma unroll
  for (int kt = 0; kt < 8; ++kt) {
    FragBF a;
    const bf16_t* ap = sA + row * 256 + kt * 32 + khalf;
    a.q[0] = *(const uv4*)ap;          // K {0..7} / {8..15}
    a.q[1] = *(const uv4*)(ap + 16);   // K {16..23} / {24..31}

    const bf16_t* bbase = w1p + (((size_t)kt * 64 + wave * 8) * 32 + lane) * 16;
    if (kt < 7)
      __builtin_prefetch(bbase + (size_t)64 * 32 * 16, 0, 1);  // next k-slab

    FragBF bb[8];
#pragma unroll
    for (int nt = 0; nt < 8; ++nt) {          // batch all B-frag loads first
      const uv4* bp = (const uv4*)(bbase + (size_t)nt * 32 * 16);
      bb[nt].q[0] = bp[0];
      bb[nt].q[1] = bp[1];
    }
#pragma unroll
    for (int nt = 0; nt < 8; ++nt)            // then 8 back-to-back WMMAs
      acc[nt] = __builtin_amdgcn_wmma_f32_16x16x32_bf16(
          false, a.v, false, bb[nt].v, (short)0, acc[nt], false, false);
  }
#pragma unroll
  for (int nt = 0; nt < 8; ++nt) {
    int ncol = wave * 128 + nt * 16 + (lane & 15);
    float bias = b1[ncol];
#pragma unroll
    for (int r = 0; r < 8; ++r) {
      float v = acc[nt][r] + bias;
      v = v > 0.f ? v : 0.2f * v;
      sFF[(rbase + r) * FF_N + ncol] = (bf16_t)v;
    }
  }
  __syncthreads();

  // 48 dot products (16 rows x 3 outputs), wave-parallel reduction
  for (int dd = wave; dd < 48; dd += 8) {
    int r = dd / 3;
    int c = dd - r * 3;
    float a = 0.f;
    for (int k = lane; k < FF_N; k += 32) a += (float)sFF[r * FF_N + k] * w2[k * 3 + c];
    a += __shfl_xor(a, 16);
    a += __shfl_xor(a, 8);
    a += __shfl_xor(a, 4);
    a += __shfl_xor(a, 2);
    a += __shfl_xor(a, 1);
    if (lane == 0) {
      float v = a + b2[c];
      out[(size_t)(m0 + r) * 3 + c] = v > 0.f ? v : 0.2f * v;
    }
  }
}

// ---------------------------------------------------------------- launcher
extern "C" void kernel_launch(void* const* d_in, const int* in_sizes, int n_in,
                              void* d_out, int out_size, void* d_ws, size_t ws_size,
                              hipStream_t stream) {
  (void)in_sizes; (void)n_in; (void)out_size; (void)ws_size;
  const float* x      = (const float*)d_in[0];
  const float* ct     = (const float*)d_in[1];
  const float* cs     = (const float*)d_in[2];
  const float* in_w1  = (const float*)d_in[3];
  const float* in_b1  = (const float*)d_in[4];
  const float* in_w2  = (const float*)d_in[5];
  const float* in_b2  = (const float*)d_in[6];
  const float* pe_w1  = (const float*)d_in[7];
  const float* pe_b1  = (const float*)d_in[8];
  const float* pe_w2  = (const float*)d_in[9];
  const float* pe_b2  = (const float*)d_in[10];
  const float* ln_g   = (const float*)d_in[11];
  const float* ln_b   = (const float*)d_in[12];
  const float* q_w    = (const float*)d_in[13];
  const float* k_w    = (const float*)d_in[14];
  const float* v_w    = (const float*)d_in[15];
  const float* lscale = (const float*)d_in[16];
  const float* out_w  = (const float*)d_in[17];
  const float* mo_w1  = (const float*)d_in[18];
  const float* mo_b1  = (const float*)d_in[19];
  const float* mo_w2  = (const float*)d_in[20];
  const float* mo_b2  = (const float*)d_in[21];
  float* out = (float*)d_out;

  char* ws = (char*)d_ws;
  size_t off = 0;
  auto suballoc = [&](size_t bytes) -> void* {
    void* p = ws + off;
    off += (bytes + 255) & ~(size_t)255;
    return p;
  };
  int*    idx_ws = (int*)   suballoc((size_t)T_N * NH_N * sizeof(int));          // 256 KB
  float*  h_ws   = (float*) suballoc((size_t)B_N * S_N * MD_N * sizeof(float));  // 8 MB
  float*  pe_ws  = (float*) suballoc((size_t)T_N * NH_N * MD_N * sizeof(float)); // 8 MB
  bf16_t* xn_ws  = (bf16_t*)suballoc((size_t)B_N * T_N * 256 * sizeof(bf16_t));  // 32 MB
  bf16_t* w1p_ws = (bf16_t*)suballoc((size_t)256 * FF_N * sizeof(bf16_t));       // 512 KB

  knn_kernel<<<T_N / 256, 256, 0, stream>>>(ct, cs, idx_ws);
  embed_kernel<<<(B_N * S_N) / 256, 256, 0, stream>>>(x, in_w1, in_b1, in_w2, in_b2, h_ws);
  pack_w1_kernel<<<(256 * FF_N) / 256, 256, 0, stream>>>(mo_w1, w1p_ws);
  pe_kernel<<<(T_N * NH_N) / 256, 256, 0, stream>>>(ct, cs, idx_ws, pe_w1, pe_b1, pe_w2, pe_b2, pe_ws);
  attn_kernel<<<(B_N * T_N) / 8, 256, 0, stream>>>(h_ws, pe_ws, idx_ws, q_w, k_w, v_w,
                                                   ln_g, ln_b, lscale, out_w, xn_ws);
  mlp_kernel<<<(B_N * T_N) / 16, 256, 0, stream>>>(xn_ws, w1p_ws, mo_b1, mo_w2, mo_b2, out);
}
